// MultiHeadedAttention_84095459655712
// MI455X (gfx1250) — matrix operand order repaired and checked by reading, not compiled, not run on hardware
//
#include <hip/hip_runtime.h>
#include <hip/hip_bf16.h>

// ---------------------------------------------------------------------------
// MultiHeadedAttention for MI455X (gfx1250), bf16 WMMA + flash attention,
// async global->LDS staging (ASYNCcnt) with double buffering in the GEMMs.
// B=2, S=2048, D=1024, H=16, DK=64, DFF=128 (t-MLP).
// ---------------------------------------------------------------------------

typedef __bf16 v16bf __attribute__((ext_vector_type(16)));
typedef float  v8f   __attribute__((ext_vector_type(8)));

#define B_   2
#define S_   2048
#define D_   1024
#define H_   16
#define DK_  64
#define M_   (B_ * S_)   // 4096
#define DFF_ 128

// fp32 -> bf16 round-to-nearest-even
static __device__ __forceinline__ unsigned short f2bf(float f) {
    unsigned int u = __builtin_bit_cast(unsigned int, f);
    u += 0x7fffu + ((u >> 16) & 1u);
    return (unsigned short)(u >> 16);
}

// Assemble a 16-element bf16 fragment from two 16-byte chunks.
static __device__ __forceinline__ v16bf ld_frag(const unsigned short* p0,
                                                const unsigned short* p1) {
    union { v16bf v; uint4 q[2]; } u;
    u.q[0] = *reinterpret_cast<const uint4*>(p0);
    u.q[1] = *reinterpret_cast<const uint4*>(p1);
    return u.v;
}

static __device__ __forceinline__ v8f vzero() {
    v8f z;
#pragma unroll
    for (int i = 0; i < 8; ++i) z[i] = 0.0f;
    return z;
}

static __device__ __forceinline__ v8f wmma_bf16(v16bf a, v16bf b, v8f c) {
#if defined(__gfx1250__)
    return __builtin_amdgcn_wmma_f32_16x16x32_bf16(
        /*neg_a=*/false, a, /*neg_b=*/false, b,
        /*c_mod=*/(short)0, c, /*reuse_a=*/false, /*reuse_b=*/false);
#else
    (void)a; (void)b;
    return c;
#endif
}

// Async global -> LDS copy of 16 bytes per lane (tracked by ASYNCcnt).
// LDS generic pointers carry the LDS byte offset in their low 32 bits.
static __device__ __forceinline__ void async_ld_b128(void* lds, const void* g) {
#if defined(__gfx1250__)
    unsigned off = (unsigned)(unsigned long long)lds;
    asm volatile("global_load_async_to_lds_b128 %0, %1, off"
                 :: "v"(off), "v"(g) : "memory");
#else
    *reinterpret_cast<uint4*>(lds) = *reinterpret_cast<const uint4*>(g);
#endif
}

static __device__ __forceinline__ void wait_async0() {
#if defined(__gfx1250__)
    asm volatile("s_wait_asynccnt 0x0" ::: "memory");
#endif
}

// ---------------------------------------------------------------------------
// fp32 -> bf16 conversion (grid-stride)
// ---------------------------------------------------------------------------
__global__ void cvt_bf16_kernel(const float* __restrict__ in,
                                unsigned short* __restrict__ out, int n) {
    int i = blockIdx.x * blockDim.x + threadIdx.x;
    int stride = gridDim.x * blockDim.x;
    for (; i < n; i += stride) out[i] = f2bf(in[i]);
}

// ---------------------------------------------------------------------------
// t-MLP: tbias[b, d] = tb2[d] + sum_f relu(t[b]*tW1[f] + tb1[f]) * tW2[d, f]
// ---------------------------------------------------------------------------
__global__ void tbias_kernel(const float* __restrict__ t,
                             const float* __restrict__ tW1,
                             const float* __restrict__ tb1,
                             const float* __restrict__ tW2,
                             const float* __restrict__ tb2,
                             float* __restrict__ tbias) {
    int b = blockIdx.x;
    int d = threadIdx.x;           // 0..63
    float tv  = t[b];
    float acc = tb2[d];
    for (int f = 0; f < DFF_; ++f) {
        float h = tv * tW1[f] + tb1[f];
        h = h > 0.0f ? h : 0.0f;
        acc += h * tW2[d * DFF_ + f];
    }
    tbias[b * DK_ + d] = acc;
}

// ---------------------------------------------------------------------------
// GEMM: C[m,n] = sum_k A[m,k] * W[n,k] + bias[n]   (A,W bf16; accumulate f32)
// mode 0: write bf16 to [B,H,S,DK]               (Q, V projections)
// mode 1: mode 0 + tbias[b, dk]                  (K projection)
// mode 2: write fp32 to [M, D] = [B,S,D]         (output projection)
// Block: 128(M) x 128(N) tile, 256 threads = 8 waves (2x4), wave = 64x32.
// Double-buffered LDS tiles filled with async global->LDS copies.
// ---------------------------------------------------------------------------
__global__ __launch_bounds__(256) void gemm_bf16_kernel(
    const unsigned short* __restrict__ A,   // [M_, D_]
    const unsigned short* __restrict__ W,   // [D_, D_] (torch [out,in])
    const float* __restrict__ bias,         // [D_]
    const float* __restrict__ tbias,        // [B_*DK_] (mode 1) or nullptr
    unsigned short* __restrict__ obf,       // bf16 out (modes 0/1)
    float* __restrict__ ofp,                // fp32 out (mode 2)
    int mode) {
    const int m0   = blockIdx.x * 128;
    const int n0   = blockIdx.y * 128;
    const int tid  = threadIdx.x;
    const int lane = tid & 31;
    const int wave = tid >> 5;
    const int wm   = (wave >> 2) * 64;      // wave m-offset (0,64)
    const int wn   = (wave & 3) * 32;       // wave n-offset (0,32,64,96)
    const int lm   = lane & 15;
    const int hi   = lane >> 4;
    const int klo  = hi * 8;                // A-frag chunk0 k-offset
    const int khi  = hi * 16;               // unused; B-frags also use klo

    __shared__ __align__(16) unsigned short At[2][128][32 + 8];
    __shared__ __align__(16) unsigned short Wt[2][128][32 + 8];

    v8f acc[4][2];
#pragma unroll
    for (int mt = 0; mt < 4; ++mt)
#pragma unroll
        for (int nt = 0; nt < 2; ++nt) acc[mt][nt] = vzero();

    const int lrow  = tid >> 1;             // 0..127
    const int lhalf = (tid & 1) * 16;       // 0 or 16
    const unsigned short* Arow = A + (size_t)(m0 + lrow) * D_;
    const unsigned short* Wrow = W + (size_t)(n0 + lrow) * D_;

    // Issue async fills of a 128x32 tile pair into buffer `buf` for k = kk.
    auto issue = [&](int buf, int kk) {
        async_ld_b128(&At[buf][lrow][lhalf],     Arow + kk + lhalf);
        async_ld_b128(&At[buf][lrow][lhalf + 8], Arow + kk + lhalf + 8);
        async_ld_b128(&Wt[buf][lrow][lhalf],     Wrow + kk + lhalf);
        async_ld_b128(&Wt[buf][lrow][lhalf + 8], Wrow + kk + lhalf + 8);
    };

    issue(0, 0);   // prologue

    for (int kk = 0; kk < D_; kk += 32) {
        const int buf = (kk >> 5) & 1;
        wait_async0();          // own tile writes for `buf` are in LDS
        __syncthreads();        // everyone's writes visible; prev compute done
        if (kk + 32 < D_) issue(buf ^ 1, kk + 32);   // overlap with WMMAs

        v16bf bfrag[2];
#pragma unroll
        for (int nt = 0; nt < 2; ++nt) {
            const unsigned short* r = &Wt[buf][wn + nt * 16 + lm][klo];
            bfrag[nt] = ld_frag(r, r + 16);
        }
#pragma unroll
        for (int mt = 0; mt < 4; ++mt) {
            const unsigned short* r = &At[buf][wm + mt * 16 + lm][klo];
            v16bf afrag = ld_frag(r, r + 16);
#pragma unroll
            for (int nt = 0; nt < 2; ++nt)
                acc[mt][nt] = wmma_bf16(afrag, bfrag[nt], acc[mt][nt]);
        }
    }

    // Epilogue. C layout: n = lm, m = r + 8*hi per VGPR r.
#pragma unroll
    for (int mt = 0; mt < 4; ++mt) {
#pragma unroll
        for (int nt = 0; nt < 2; ++nt) {
            const int n  = n0 + wn + nt * 16 + lm;
            const float bn = bias[n];
#pragma unroll
            for (int r = 0; r < 8; ++r) {
                const int m = m0 + wm + mt * 16 + r + hi * 8;
                float v = acc[mt][nt][r] + bn;
                if (mode == 2) {
                    ofp[(size_t)m * D_ + n] = v;
                } else {
                    const int b = m >> 11, s = m & (S_ - 1);
                    const int h = n >> 6,  dk = n & 63;
                    if (mode == 1) v += tbias[b * DK_ + dk];
                    obf[(((size_t)(b * H_ + h)) * S_ + s) * DK_ + dk] = f2bf(v);
                }
            }
        }
    }
}

// ---------------------------------------------------------------------------
// Fused flash attention. Grid (B*H, S/128), 256 threads = 8 waves.
// Wave w owns query rows [q0+16w, q0+16w+16). Q frags live in registers.
// K tile [128 keys][64 dk] staged with async copies; V tile transposed
// through registers into [64 dk][128 keys]. Online softmax; P goes through
// wave-private LDS for the C->A layout change. Next tiles are prefetched.
// Output written as bf16 [B, S, H*DK] (row-major) to feed the Wo GEMM.
// ---------------------------------------------------------------------------
__global__ __launch_bounds__(256) void attn_kernel(
    const unsigned short* __restrict__ Q,   // [B,H,S,DK]
    const unsigned short* __restrict__ K,   // [B,H,S,DK] (t-bias already added)
    const unsigned short* __restrict__ V,   // [B,H,S,DK]
    unsigned short* __restrict__ O) {       // [B,S,H*DK] bf16
    const int bh  = blockIdx.x;             // 0..B*H-1
    const int qb  = blockIdx.y;             // 0..S/128-1
    const int b   = bh >> 4;
    const int h   = bh & 15;
    const size_t base = (size_t)bh * S_ * DK_;

    const int tid  = threadIdx.x;
    const int lane = tid & 31;
    const int wave = tid >> 5;
    const int lm   = lane & 15;
    const int hi   = lane >> 4;
    const int klo  = hi * 8;
    const int khi  = hi * 16;

    __shared__ __align__(16) unsigned short Kt[128][64];      // 16 KB
    __shared__ __align__(16) unsigned short Vt[64][128];      // 16 KB (transposed)
    __shared__ __align__(16) unsigned short Pb[8][16][128];   // 32 KB, per-wave P

    // Q fragments (16 rows x 64 dk) -> two A-frags, loaded once.
    const int q0 = qb * 128 + wave * 16;
    const unsigned short* qrow = Q + base + (size_t)(q0 + lm) * DK_;
    const v16bf aq0 = ld_frag(qrow + klo,      qrow + 16 + klo);
    const v16bf aq1 = ld_frag(qrow + 32 + klo, qrow + 48 + klo);

    float mrow[8], lrow[8];
    v8f o[4];
#pragma unroll
    for (int r = 0; r < 8; ++r) { mrow[r] = -1e30f; lrow[r] = 0.0f; }
#pragma unroll
    for (int dt = 0; dt < 4; ++dt) o[dt] = vzero();

    for (int kb = 0; kb < S_ / 128; ++kb) {
        // Stage K tile (async -> LDS) + transposed V tile (each thread 4x16B).
#pragma unroll
        for (int i = 0; i < 4; ++i) {
            const int u   = tid + i * 256;        // 0..1023
            const int row = u >> 3;               // key 0..127
            const int c8  = (u & 7) * 8;          // dk chunk base
            const size_t g = base + (size_t)(kb * 128 + row) * DK_ + c8;
            async_ld_b128(&Kt[row][c8], K + g);
            uint4 vv = *reinterpret_cast<const uint4*>(V + g);
            const unsigned short* pv = reinterpret_cast<const unsigned short*>(&vv);
#pragma unroll
            for (int j = 0; j < 8; ++j) Vt[c8 + j][row] = pv[j];
        }
        // Prefetch next key block while this one is consumed.
        if (kb + 1 < S_ / 128) {
            const size_t gn = base + (size_t)(kb + 1) * 128 * DK_ + (size_t)tid * 32;
            __builtin_prefetch(K + gn, 0, 1);
            __builtin_prefetch(V + gn, 0, 1);
        }
        wait_async0();
        __syncthreads();

        // Scores: S[16 x 128] = Q x K^T  (8 n-tiles, 2 k-steps each).
        v8f sc[8];
#pragma unroll
        for (int nt = 0; nt < 8; ++nt) {
            const unsigned short* r0 = &Kt[nt * 16 + lm][klo];
            const unsigned short* r1 = &Kt[nt * 16 + lm][32 + klo];
            v8f c = vzero();
            c = wmma_bf16(aq0, ld_frag(r0, r0 + 16), c);
            c = wmma_bf16(aq1, ld_frag(r1, r1 + 16), c);
            sc[nt] = c;
        }
#pragma unroll
        for (int nt = 0; nt < 8; ++nt)
#pragma unroll
            for (int e = 0; e < 8; ++e) sc[nt][e] *= 0.125f;   // 1/sqrt(DK)

        // Online softmax per row (row m = r + 8*hi lives in a 16-lane half).
#pragma unroll
        for (int r = 0; r < 8; ++r) {
            float tm = -1e30f;
#pragma unroll
            for (int nt = 0; nt < 8; ++nt) tm = fmaxf(tm, sc[nt][r]);
            tm = fmaxf(tm, __shfl_xor(tm, 1));
            tm = fmaxf(tm, __shfl_xor(tm, 2));
            tm = fmaxf(tm, __shfl_xor(tm, 4));
            tm = fmaxf(tm, __shfl_xor(tm, 8));
            const float mnew  = fmaxf(mrow[r], tm);
            const float alpha = __expf(mrow[r] - mnew);
            float psum = 0.0f;
#pragma unroll
            for (int nt = 0; nt < 8; ++nt) {
                const float p = __expf(sc[nt][r] - mnew);
                psum += p;
                Pb[wave][r + hi * 8][nt * 16 + lm] = f2bf(p);
            }
            psum += __shfl_xor(psum, 1);
            psum += __shfl_xor(psum, 2);
            psum += __shfl_xor(psum, 4);
            psum += __shfl_xor(psum, 8);
            lrow[r] = lrow[r] * alpha + psum;
            mrow[r] = mnew;
#pragma unroll
            for (int dt = 0; dt < 4; ++dt) o[dt][r] *= alpha;
        }

        // O += P x V   (4 key-chunks of 32, 4 dk-tiles of 16).
#pragma unroll
        for (int kc = 0; kc < 4; ++kc) {
            const unsigned short* pr = &Pb[wave][lm][kc * 32 + klo];
            const v16bf ap = ld_frag(pr, pr + 16);
#pragma unroll
            for (int dt = 0; dt < 4; ++dt) {
                const unsigned short* vr = &Vt[dt * 16 + lm][kc * 32 + klo];
                o[dt] = wmma_bf16(ap, ld_frag(vr, vr + 16), o[dt]);
            }
        }
        __syncthreads();
    }

    // Normalize and emit bf16 rows of [B,S,H*DK].
#pragma unroll
    for (int r = 0; r < 8; ++r) {
        const float inv = 1.0f / lrow[r];
        const int   row = q0 + r + hi * 8;
#pragma unroll
        for (int dt = 0; dt < 4; ++dt) {
            const float v = o[dt][r] * inv;
            O[((size_t)(b * S_) + row) * D_ + h * DK_ + dt * 16 + lm] = f2bf(v);
        }
    }
}

// ---------------------------------------------------------------------------
// Host-side orchestration
// ---------------------------------------------------------------------------
extern "C" void kernel_launch(void* const* d_in, const int* in_sizes, int n_in,
                              void* d_out, int out_size, void* d_ws, size_t ws_size,
                              hipStream_t stream) {
    (void)in_sizes; (void)n_in; (void)out_size; (void)ws_size;

    const float* q   = (const float*)d_in[0];
    const float* k   = (const float*)d_in[1];
    const float* v   = (const float*)d_in[2];
    const float* t   = (const float*)d_in[3];
    const float* Wq  = (const float*)d_in[4];
    const float* bq  = (const float*)d_in[5];
    const float* Wk  = (const float*)d_in[6];
    const float* bk  = (const float*)d_in[7];
    const float* Wv  = (const float*)d_in[8];
    const float* bv  = (const float*)d_in[9];
    const float* Wo  = (const float*)d_in[10];
    const float* bo  = (const float*)d_in[11];
    const float* tW1 = (const float*)d_in[12];
    const float* tb1 = (const float*)d_in[13];
    const float* tW2 = (const float*)d_in[14];
    const float* tb2 = (const float*)d_in[15];
    float* out = (float*)d_out;

    char* ws = (char*)d_ws;
    size_t off = 0;
    auto alloc = [&](size_t bytes) -> void* {
        void* p = ws + off;
        off += (bytes + 255) & ~(size_t)255;
        return p;
    };
    unsigned short* xq  = (unsigned short*)alloc((size_t)M_ * D_ * 2);
    unsigned short* xk  = (unsigned short*)alloc((size_t)M_ * D_ * 2);
    unsigned short* xv  = (unsigned short*)alloc((size_t)M_ * D_ * 2);
    unsigned short* wqb = (unsigned short*)alloc((size_t)D_ * D_ * 2);
    unsigned short* wkb = (unsigned short*)alloc((size_t)D_ * D_ * 2);
    unsigned short* wvb = (unsigned short*)alloc((size_t)D_ * D_ * 2);
    unsigned short* wob = (unsigned short*)alloc((size_t)D_ * D_ * 2);
    unsigned short* Qb  = (unsigned short*)alloc((size_t)M_ * D_ * 2);
    unsigned short* Kb  = (unsigned short*)alloc((size_t)M_ * D_ * 2);
    unsigned short* Vb  = (unsigned short*)alloc((size_t)M_ * D_ * 2);
    unsigned short* Ab  = (unsigned short*)alloc((size_t)M_ * D_ * 2);
    float*          tbs = (float*)alloc((size_t)B_ * DK_ * 4);

    const int nAct = M_ * D_;   // 4 Mi elements
    const int nW   = D_ * D_;   // 1 Mi elements
    cvt_bf16_kernel<<<2048, 256, 0, stream>>>(q,  xq,  nAct);
    cvt_bf16_kernel<<<2048, 256, 0, stream>>>(k,  xk,  nAct);
    cvt_bf16_kernel<<<2048, 256, 0, stream>>>(v,  xv,  nAct);
    cvt_bf16_kernel<<<1024, 256, 0, stream>>>(Wq, wqb, nW);
    cvt_bf16_kernel<<<1024, 256, 0, stream>>>(Wk, wkb, nW);
    cvt_bf16_kernel<<<1024, 256, 0, stream>>>(Wv, wvb, nW);
    cvt_bf16_kernel<<<1024, 256, 0, stream>>>(Wo, wob, nW);

    tbias_kernel<<<B_, DK_, 0, stream>>>(t, tW1, tb1, tW2, tb2, tbs);

    dim3 gg(M_ / 128, D_ / 128);   // (32, 8)
    gemm_bf16_kernel<<<gg, 256, 0, stream>>>(xq, wqb, bq, nullptr, Qb, nullptr, 0);
    gemm_bf16_kernel<<<gg, 256, 0, stream>>>(xk, wkb, bk, tbs,     Kb, nullptr, 1);
    gemm_bf16_kernel<<<gg, 256, 0, stream>>>(xv, wvb, bv, nullptr, Vb, nullptr, 0);

    dim3 ga(B_ * H_, S_ / 128);    // (32, 16)
    attn_kernel<<<ga, 256, 0, stream>>>(Qb, Kb, Vb, Ab);

    gemm_bf16_kernel<<<gg, 256, 0, stream>>>(Ab, wob, bo, nullptr, nullptr, out, 2);
}
